// CustomConv2d_20942260535369
// MI455X (gfx1250) — compile-verified, matrix operand order checked by reading, not run on hardware
//
#include <hip/hip_runtime.h>

typedef __attribute__((ext_vector_type(16))) _Float16 v16h;
typedef __attribute__((ext_vector_type(8)))  _Float16 v8h;
typedef __attribute__((ext_vector_type(8)))  float    v8f;
typedef __attribute__((ext_vector_type(4)))  float    v4f;
typedef __attribute__((ext_vector_type(4)))  unsigned int u32x4;
typedef __attribute__((ext_vector_type(8)))  int      i32x8;
typedef __attribute__((ext_vector_type(4)))  int      i32x4;

namespace {
constexpr int B_    = 32;
constexpr int CIN   = 128;
constexpr int HW    = 112;
constexpr int COUT  = 256;
constexpr int NPIX  = HW * HW;     // 12544
constexpr int KDIM  = CIN * 9;     // 1152, reordered as k = (kh*3+kw)*128 + ci
constexpr int MTILE = 128;
constexpr int NTILE = 128;
constexpr int KTILE = 32;
constexpr int NCH   = KDIM / KTILE;  // 36 chunks
constexpr int ASTR  = 40;          // 32 data f16 + 8 pad f16 (64B data + 16B pad)
constexpr int BSTR  = 40;
}

#if defined(__has_builtin)
#if __has_builtin(__builtin_amdgcn_tensor_load_to_lds) && \
    __has_builtin(__builtin_amdgcn_s_wait_tensorcnt)
#define HAVE_TDM 1
#endif
#endif
#ifndef HAVE_TDM
#define HAVE_TDM 0
#endif

// ---------------------------------------------------------------------------
// Pass 1: repack weights OIHW fp32 -> f16, K reordered to (kh*3+kw)*128 + ci.
// ---------------------------------------------------------------------------
__global__ __launch_bounds__(256)
void repack_weights(const float* __restrict__ w, _Float16* __restrict__ wt) {
  const int idx = blockIdx.x * 256 + threadIdx.x;   // 256*1152 total
  const int m  = idx / KDIM;
  const int t  = idx - m * KDIM;
  const int r  = t >> 7;           // kh*3+kw
  const int ci = t & 127;
  wt[idx] = (_Float16)w[(size_t)m * KDIM + ci * 9 + r];
}

// ---------------------------------------------------------------------------
// Pass 2: implicit GEMM, M=256, N=B*H*W, K=1152 (tap-major), f16 WMMA.
// Double-buffered: TDM streams A tiles, x loads prefetched into registers.
// ---------------------------------------------------------------------------
__global__ __launch_bounds__(256)
void conv3x3_gemm_tdm(const float* __restrict__ x,
                      const _Float16* __restrict__ wt,
                      const float* __restrict__ bias,
                      float* __restrict__ out)
{
  __shared__ alignas(16) _Float16 Abuf[2][MTILE * ASTR];   // weights [m][k]
  __shared__ alignas(16) _Float16 Bbuf[2][NTILE * BSTR];   // im2col [p][k]

  const int tid  = threadIdx.x;
  const int lane = tid & 31;
  const int wave = tid >> 5;
  const int lh   = lane >> 4;
  const int lr   = lane & 15;

  const int m0  = blockIdx.y * MTILE;
  const int p0  = blockIdx.x * NTILE;
  const int img = p0 / NPIX;                 // tile never crosses an image

  // fixed per-thread pixel for B staging; each thread owns 16 contiguous k
  const int pcol  = tid & 127;
  const int sl    = p0 + pcol - img * NPIX;
  const int oh    = sl / HW;
  const int ow    = sl - oh * HW;
  const int khalf = tid >> 7;                // k sub-range 0..15 / 16..31

  // per-thread row/half for (non-TDM) A staging
  const int am    = tid >> 1;
  const int ahalf = tid & 1;

  const int wave_m = wave & 1;
  const int wave_n = wave >> 1;

  v8f acc[4][2] = {};
  float breg[16];
  bool  bvalid = true;

  const float* xb = x + (size_t)img * CIN * NPIX;

  // ---- A stage: TDM 2D tile copy Wt[m0..+127][k0..+31] -> Abuf[buf]
  auto stageA = [&](int k0, int buf) {
#if HAVE_TDM
    if (wave == 0) {
      const unsigned long long ga =
          (unsigned long long)(const void*)(wt + (size_t)m0 * KDIM + k0);
      const unsigned ldsA = (unsigned)(size_t)(void*)&Abuf[buf][0];
      u32x4 g0 = { 1u,                                   // count=1 valid D#
                   ldsA,                                 // lds_addr
                   (unsigned)ga,                         // global_addr[31:0]
                   ((unsigned)(ga >> 32) & 0x01FFFFFFu) | 0x80000000u }; // type=2
      i32x8 g1 = { (int)0x06D10000,      // 2B elems, pad_en, every 16dw pad 4dw
                   (int)(KDIM << 16),    // tensor_dim0 = 1152 (low16)
                   (int)(COUT << 16),    // tensor_dim1 = 256
                   (int)(KTILE << 16),   // tile_dim0 = 32
                   (int)MTILE,           // tile_dim1 = 128
                   (int)KDIM,            // tensor_dim0_stride = 1152
                   0, 0 };
      i32x4 z4 = { 0, 0, 0, 0 };
      i32x8 z8 = { 0, 0, 0, 0, 0, 0, 0, 0 };
      __builtin_amdgcn_tensor_load_to_lds(g0, g1, z4, z4, z8, 0);
    }
#else
    {
      const _Float16* src = wt + (size_t)(m0 + am) * KDIM + k0 + ahalf * 16;
      v8h h0 = *(const v8h*)src;
      v8h h1 = *(const v8h*)(src + 8);
      _Float16* dst = &Abuf[buf][am * ASTR + ahalf * 16];
      *(v8h*)dst       = h0;
      *(v8h*)(dst + 8) = h1;
    }
#endif
  };

  // ---- B stage, part 1: UNCONDITIONAL global loads (clamped address) so the
  //      compiler issues one clause and sinks the wait past the WMMAs.
  auto loadB = [&](int k0) {
    const int r   = k0 >> 7;                 // fixed filter tap for chunk
    const int cib = (k0 & 127) + khalf * 16; // this thread's channel base
    const int kh = r / 3, kw = r - kh * 3;
    const int ih = oh + kh - 1;
    const int iw = ow + kw - 1;
    bvalid = (unsigned)ih < (unsigned)HW && (unsigned)iw < (unsigned)HW;
    const int off = bvalid ? (ih * HW + iw) : 0;   // always in-bounds
    const float* src = xb + (size_t)cib * NPIX + off;
    #pragma unroll
    for (int j = 0; j < 16; ++j)
      breg[j] = src[(size_t)j * NPIX];
  };

  // ---- B stage, part 2: zero-select + cvt + two ds_store_b128
  auto storeB = [&](int buf) {
    v8h h0, h1;
    #pragma unroll
    for (int j = 0; j < 8; ++j) {
      h0[j] = (_Float16)(bvalid ? breg[j]     : 0.f);
      h1[j] = (_Float16)(bvalid ? breg[8 + j] : 0.f);
    }
    _Float16* dst = &Bbuf[buf][pcol * BSTR + khalf * 16];
    *(v8h*)dst       = h0;
    *(v8h*)(dst + 8) = h1;
  };

  // ---- prologue: fill buffer 0
  stageA(0, 0);
  loadB(0);
  storeB(0);
#if HAVE_TDM
  if (wave == 0) __builtin_amdgcn_s_wait_tensorcnt(0);
#endif
  __syncthreads();

  for (int c = 0; c < NCH; ++c) {
    const int cur = c & 1;

    // kick off next chunk: DMA for A, register prefetch for B
    if (c + 1 < NCH) {
      stageA((c + 1) * KTILE, cur ^ 1);
      loadB((c + 1) * KTILE);
    }

    // ---- compute on current buffers
    v16h afrag[4], bfrag[2];
    #pragma unroll
    for (int i = 0; i < 4; ++i) {
      const _Float16* ap = &Abuf[cur][(wave_m * 64 + i * 16 + lr) * ASTR + lh * 8];
      v8h lo = *(const v8h*)ap;
      v8h hi = *(const v8h*)(ap + 16);
      afrag[i] = __builtin_shufflevector(lo, hi,
                   0,1,2,3,4,5,6,7,8,9,10,11,12,13,14,15);
    }
    #pragma unroll
    for (int j = 0; j < 2; ++j) {
      const _Float16* bp = &Bbuf[cur][(wave_n * 32 + j * 16 + lr) * BSTR + lh * 8];
      v8h lo = *(const v8h*)bp;
      v8h hi = *(const v8h*)(bp + 16);
      bfrag[j] = __builtin_shufflevector(lo, hi,
                   0,1,2,3,4,5,6,7,8,9,10,11,12,13,14,15);
    }
    #pragma unroll
    for (int i = 0; i < 4; ++i)
      #pragma unroll
      for (int j = 0; j < 2; ++j)
        acc[i][j] = __builtin_amdgcn_wmma_f32_16x16x32_f16(
            false, afrag[i], false, bfrag[j], (short)0, acc[i][j], false, false);

    // publish next B tile after compute (x loads had the WMMAs to land)
    if (c + 1 < NCH) storeB(cur ^ 1);
#if HAVE_TDM
    if (wave == 0) __builtin_amdgcn_s_wait_tensorcnt(0);
#endif
    __syncthreads();
  }

  // ---- epilogue
  const size_t obase = (size_t)img * COUT * NPIX;
  const int prow = p0 - img * NPIX + wave_n * 32 + lr;
  #pragma unroll
  for (int i = 0; i < 4; ++i) {
    #pragma unroll
    for (int rr = 0; rr < 8; ++rr) {
      const int co = m0 + wave_m * 64 + i * 16 + lh * 8 + rr;
      const float bv = bias[co];
      #pragma unroll
      for (int j = 0; j < 2; ++j)
        out[obase + (size_t)co * NPIX + prow + j * 16] = acc[i][j][rr] + bv;
    }
  }
}

// ---------------------------------------------------------------------------
// Fallback (no workspace): fp32 sources, on-the-fly cvt.
// ---------------------------------------------------------------------------
__global__ __launch_bounds__(256)
void conv3x3_fallback(const float* __restrict__ x,
                      const float* __restrict__ w,
                      const float* __restrict__ bias,
                      float* __restrict__ out)
{
  __shared__ alignas(16) _Float16 Abuf[MTILE * ASTR];
  __shared__ alignas(16) _Float16 Bbuf[NTILE * BSTR];

  const int tid  = threadIdx.x;
  const int lane = tid & 31;
  const int wave = tid >> 5;
  const int lh   = lane >> 4;
  const int lr   = lane & 15;

  const int m0  = blockIdx.y * MTILE;
  const int p0  = blockIdx.x * NTILE;
  const int img = p0 / NPIX;

  const int pcol  = tid & 127;
  const int sl    = p0 + pcol - img * NPIX;
  const int oh    = sl / HW;
  const int ow    = sl - oh * HW;
  const int khalf = tid >> 7;

  const int am    = tid >> 1;
  const int ahalf = tid & 1;

  const int wave_m = wave & 1;
  const int wave_n = wave >> 1;

  v8f acc[4][2] = {};

  const float* wrow = w + (size_t)(m0 + am) * KDIM + ahalf * 16;
  const float* xb   = x + (size_t)img * CIN * NPIX;

  for (int k0 = 0; k0 < KDIM; k0 += KTILE) {
    {
      v4f f0 = *(const v4f*)(wrow + k0);
      v4f f1 = *(const v4f*)(wrow + k0 + 4);
      v4f f2 = *(const v4f*)(wrow + k0 + 8);
      v4f f3 = *(const v4f*)(wrow + k0 + 12);
      v8h h0, h1;
      #pragma unroll
      for (int i = 0; i < 4; ++i) {
        h0[i] = (_Float16)f0[i]; h0[i+4] = (_Float16)f1[i];
        h1[i] = (_Float16)f2[i]; h1[i+4] = (_Float16)f3[i];
      }
      _Float16* dst = &Abuf[am * ASTR + ahalf * 16];
      *(v8h*)dst = h0; *(v8h*)(dst + 8) = h1;
    }
    #pragma unroll 4
    for (int j = 0; j < 16; ++j) {
      int kk = j * 2 + khalf;
      int k  = k0 + kk;
      int ci = k / 9;
      int r9 = k - ci * 9;
      int kh = r9 / 3;
      int kw = r9 - kh * 3;
      int ih = oh + kh - 1;
      int iw = ow + kw - 1;
      bool valid = (unsigned)ih < (unsigned)HW && (unsigned)iw < (unsigned)HW;
      int off = valid ? (ih * HW + iw) : 0;        // clamped, always in-bounds
      float v = xb[(size_t)ci * NPIX + off];
      Bbuf[pcol * BSTR + kk] = (_Float16)(valid ? v : 0.f);
    }
    __syncthreads();

    v16h afrag[4], bfrag[2];
    #pragma unroll
    for (int i = 0; i < 4; ++i) {
      const _Float16* ap = &Abuf[(wave_m * 64 + i * 16 + lr) * ASTR + lh * 8];
      v8h lo = *(const v8h*)ap;
      v8h hi = *(const v8h*)(ap + 16);
      afrag[i] = __builtin_shufflevector(lo, hi,
                   0,1,2,3,4,5,6,7,8,9,10,11,12,13,14,15);
    }
    #pragma unroll
    for (int j = 0; j < 2; ++j) {
      const _Float16* bp = &Bbuf[(wave_n * 32 + j * 16 + lr) * BSTR + lh * 8];
      v8h lo = *(const v8h*)bp;
      v8h hi = *(const v8h*)(bp + 16);
      bfrag[j] = __builtin_shufflevector(lo, hi,
                   0,1,2,3,4,5,6,7,8,9,10,11,12,13,14,15);
    }
    #pragma unroll
    for (int i = 0; i < 4; ++i)
      #pragma unroll
      for (int j = 0; j < 2; ++j)
        acc[i][j] = __builtin_amdgcn_wmma_f32_16x16x32_f16(
            false, afrag[i], false, bfrag[j], (short)0, acc[i][j], false, false);
    __syncthreads();
  }

  const size_t obase = (size_t)img * COUT * NPIX;
  const int prow = p0 - img * NPIX + wave_n * 32 + lr;
  #pragma unroll
  for (int i = 0; i < 4; ++i) {
    #pragma unroll
    for (int rr = 0; rr < 8; ++rr) {
      const int co = m0 + wave_m * 64 + i * 16 + lh * 8 + rr;
      const float bv = bias[co];
      #pragma unroll
      for (int j = 0; j < 2; ++j)
        out[obase + (size_t)co * NPIX + prow + j * 16] = acc[i][j][rr] + bv;
    }
  }
}

extern "C" void kernel_launch(void* const* d_in, const int* in_sizes, int n_in,
                              void* d_out, int out_size, void* d_ws, size_t ws_size,
                              hipStream_t stream) {
  (void)in_sizes; (void)n_in; (void)out_size;
  const float* x    = (const float*)d_in[0];
  const float* w    = (const float*)d_in[1];
  const float* bias = (const float*)d_in[2];
  float* out = (float*)d_out;

  const dim3 grid((B_ * NPIX) / NTILE, COUT / MTILE);   // 3136 x 2
  const size_t wbytes = (size_t)COUT * KDIM * sizeof(_Float16);   // 576 KB

  if (ws_size >= wbytes) {
    _Float16* wt = (_Float16*)d_ws;
    repack_weights<<<dim3((COUT * KDIM) / 256), dim3(256), 0, stream>>>(w, wt);
    conv3x3_gemm_tdm<<<grid, dim3(256), 0, stream>>>(x, wt, bias, out);
  } else {
    conv3x3_fallback<<<grid, dim3(256), 0, stream>>>(x, w, bias, out);
  }
}
